// ZonedBrainModel_962072674937
// MI455X (gfx1250) — matrix operand order repaired and checked by reading, not compiled, not run on hardware
//
#include <hip/hip_runtime.h>
#include <hip/hip_bf16.h>

#define USE_ASYNC_LDS 1   // global_load_async_to_lds_b128 staging (CDNA5)

// ---------------- model constants ----------------
constexpr int kB    = 8;
constexpr int kSEQ  = 2048;
constexpr int kV    = 50257;
constexpr int kD    = 128;
constexpr int kZ    = 5;
constexpr int kN    = 2048;   // zone neurons
constexpr int kH    = 1024;   // hub neurons
constexpr int kNB   = 128;    // buffer neurons
constexpr int kINJ  = 32;
constexpr int kWAVES = 5;
constexpr float kDECAY = 0.9f;

typedef __attribute__((ext_vector_type(16))) __bf16       v16bf;
typedef __attribute__((ext_vector_type(8)))  float        v8f;
typedef __attribute__((ext_vector_type(4)))  unsigned int u32x4;

union Frag16 { v16bf v; u32x4 q[2]; };

__device__ __forceinline__ v8f v8f_zero() {
  v8f r;
#pragma unroll
  for (int i = 0; i < 8; ++i) r[i] = 0.f;
  return r;
}

// ---------------------------------------------------------------------------
// Core WMMA tile:  out(128x128 strip of M x 128) = decay*old + tanh(A@B + add)
//   A  : M x K bf16 row-major (row stride K) -- per-lane register fragments
//   Bt : 128 x K bf16 (activations transposed) -- staged through LDS, shared
//        by all 8 waves of the block, double-buffered 64-K stages.
//   256 threads = 8 waves; wave w -> rows [m0 + 16w, m0 + 16w + 16)
// ---------------------------------------------------------------------------
#define LROW 9   // LDS row stride in u32x4 (144 B): conflict-free frag reads

__device__ __forceinline__ void gemm128_tile(
    const __bf16* __restrict__ A,
    const __bf16* __restrict__ Bt,
    const float*  __restrict__ Sold,
    const float*  __restrict__ addv,   // length 128 or nullptr
    float*        __restrict__ Sout,
    int K, int m0, float decay)
{
  __shared__ u32x4 lbuf[2][128 * LROW];          // 2 x 18 KB stages

  const int tid   = threadIdx.x;
  const int wave  = tid >> 5;
  const int lane  = tid & 31;
  const int lhalf = lane >> 4;   // 0/1
  const int l16   = lane & 15;

  v8f acc[8];
#pragma unroll
  for (int j = 0; j < 8; ++j) acc[j] = v8f_zero();

  const __bf16* __restrict__ arow = A + (size_t)(m0 + wave * 16 + l16) * K;

  // A fragment (16x32 bf16, ISA layout: lanes0-15 K 0-7/16-23, lanes16-31 +8)
  auto loadA = [&](int k) {
    Frag16 f;
    f.q[0] = *(const u32x4*)(arow + k +      8 * lhalf);
    f.q[1] = *(const u32x4*)(arow + k + 16 + 8 * lhalf);
    return f;
  };

#if USE_ASYNC_LDS
  // direct global -> LDS DMA staging of one 64-K B panel (ASYNCcnt-tracked)
  const unsigned ldsbase = (unsigned)(uintptr_t)(&lbuf[0][0]);
  auto async_stage = [&](int buf, int k0) {
#pragma unroll
    for (int p = 0; p < 4; ++p) {
      const int i = p * 256 + tid;
      const int n = i >> 3, q = i & 7;
      const __bf16* ga = Bt + (size_t)n * K + k0 + q * 8;
      const unsigned la =
          ldsbase + (unsigned)(buf * (128 * LROW * 16) + (n * LROW + q) * 16);
      asm volatile("global_load_async_to_lds_b128 %0, %1, off"
                   :: "v"(la), "v"(ga) : "memory");
    }
  };
#else
  auto load_stage_regs = [&](int k0, u32x4* g) {
#pragma unroll
    for (int p = 0; p < 4; ++p) {
      const int i = p * 256 + tid;
      const int n = i >> 3, q = i & 7;
      g[p] = *(const u32x4*)(Bt + (size_t)n * K + k0 + q * 8);
    }
  };
  auto store_stage = [&](int buf, const u32x4* g) {
#pragma unroll
    for (int p = 0; p < 4; ++p) {
      const int i = p * 256 + tid;
      const int n = i >> 3, q = i & 7;
      lbuf[buf][n * LROW + q] = g[p];
    }
  };
#endif

  // ---- prologue: stage 0 into LDS, first two A fragments ----
#if USE_ASYNC_LDS
  async_stage(0, 0);
  Frag16 A0 = loadA(0), A1 = loadA(32);
  asm volatile("s_wait_asynccnt 0" ::: "memory");
  __syncthreads();
#else
  {
    u32x4 g[4];
    load_stage_regs(0, g);
    store_stage(0, g);
  }
  Frag16 A0 = loadA(0), A1 = loadA(32);
  __syncthreads();
#endif

  // ---- main pipelined loop: 64 K-elements per stage ----
  for (int k0 = 0; k0 < K; k0 += 64) {
    const int  cur  = (k0 >> 6) & 1;
    const bool more = (k0 + 64) < K;
#if USE_ASYNC_LDS
    if (more) async_stage(cur ^ 1, k0 + 64);   // DMA next stage during WMMAs
#else
    u32x4 g[4];
    if (more) load_stage_regs(k0 + 64, g);
#endif
    __builtin_prefetch((const void*)(arow + k0 + 128), 0, 1);

    const u32x4* lb = &lbuf[cur][0];
    // sub-step 0 (K = k0..k0+31) with A0
#pragma unroll
    for (int jj = 0; jj < 8; jj += 4) {
      Frag16 bf[4];
#pragma unroll
      for (int j = 0; j < 4; ++j) {
        const int row = (jj + j) * 16 + l16;
        bf[j].q[0] = lb[row * LROW + lhalf];
        bf[j].q[1] = lb[row * LROW + 2 + lhalf];
      }
#pragma unroll
      for (int j = 0; j < 4; ++j)
        acc[jj + j] = __builtin_amdgcn_wmma_f32_16x16x32_bf16(
            false, A0.v, false, bf[j].v, (short)0, acc[jj + j], false, false);
    }
    // sub-step 1 (K = k0+32..k0+63) with A1
#pragma unroll
    for (int jj = 0; jj < 8; jj += 4) {
      Frag16 bf[4];
#pragma unroll
      for (int j = 0; j < 4; ++j) {
        const int row = (jj + j) * 16 + l16;
        bf[j].q[0] = lb[row * LROW + 4 + lhalf];
        bf[j].q[1] = lb[row * LROW + 6 + lhalf];
      }
#pragma unroll
      for (int j = 0; j < 4; ++j)
        acc[jj + j] = __builtin_amdgcn_wmma_f32_16x16x32_bf16(
            false, A1.v, false, bf[j].v, (short)0, acc[jj + j], false, false);
    }
    if (more) { A0 = loadA(k0 + 64); A1 = loadA(k0 + 96); }
#if USE_ASYNC_LDS
    asm volatile("s_wait_asynccnt 0" ::: "memory");
#else
    if (more) store_stage(cur ^ 1, g);
#endif
    __syncthreads();   // next stage visible; reads of cur done
  }

  // ---- epilogue: C layout -> m = m0+16w + 8*lhalf + r, n = 16j + l16 ----
#pragma unroll
  for (int j = 0; j < 8; ++j) {
    const int n = j * 16 + l16;
    const float av = addv ? addv[n] : 0.f;
#pragma unroll
    for (int r = 0; r < 8; ++r) {
      const int m = m0 + wave * 16 + lhalf * 8 + r;
      const size_t idx = (size_t)m * 128 + n;
      Sout[idx] = decay * Sold[idx] + tanhf(acc[j][r] + av);
    }
  }
}

// states[z,b] : N x 128   <- decay*states + tanh(Wz[z] @ states[z,b])
__global__ __launch_bounds__(256) void k_zone_gemm(
    const __bf16* __restrict__ Wzb, const __bf16* __restrict__ sT,
    float* __restrict__ S)
{
  const int z = blockIdx.z, b = blockIdx.y, mt = blockIdx.x;
  const __bf16* A  = Wzb + (size_t)z * kN * kN;
  const __bf16* Bt = sT  + ((size_t)(z * kB + b)) * kD * kN;
  float* Sm        = S   + ((size_t)(z * kB + b)) * kN * kD;
  gemm128_tile(A, Bt, Sm, nullptr, Sm, kN, mt * 128, kDECAY);
}

// hub[b] : H x 128  <- decay*hub + tanh(Wh @ hub[b] + inj[b] broadcast)
__global__ __launch_bounds__(256) void k_hub_gemm(
    const __bf16* __restrict__ Whb, const __bf16* __restrict__ hT,
    const float* __restrict__ inj, float* __restrict__ Hst)
{
  const int b = blockIdx.y, mt = blockIdx.x;
  const __bf16* Bt = hT  + (size_t)b * kD * kH;
  float* Hm        = Hst + (size_t)b * kH * kD;
  gemm128_tile(Whb, Bt, Hm, inj + b * kD, Hm, kH, mt * 128, kDECAY);
}

// ---------------------------------------------------------------------------
// small kernels
// ---------------------------------------------------------------------------
__global__ void k_f2bf(const float* __restrict__ src, __bf16* __restrict__ dst,
                       size_t n)
{
  size_t i = (size_t)blockIdx.x * 256 + threadIdx.x;
  if (i < n) dst[i] = (__bf16)src[i];
}

__global__ void k_embed(const int* __restrict__ ids,
                        const float* __restrict__ tok,
                        const float* __restrict__ pos,
                        float* __restrict__ emb)
{
  const int b = blockIdx.y, s = blockIdx.x, d = threadIdx.x;
  const int t = ids[b * kSEQ + s];
  emb[((size_t)b * kSEQ + s) * kD + d] =
      tok[(size_t)t * kD + d] + pos[(size_t)s * kD + d];
}

// proj[b, p] = bc[p] + last[b,:] @ Wc[:,p];  last staged in LDS (128 KB)
__global__ __launch_bounds__(256) void k_inject(
    const float* __restrict__ emb, const float* __restrict__ Wc,
    const float* __restrict__ bc, float* __restrict__ proj)
{
  extern __shared__ float lsh[];             // kB * 4096 floats
  const int t = threadIdx.x;
  for (int i = t; i < kB * kINJ * kD; i += 256) {
    const int b = i >> 12, k = i & 4095;
    lsh[i] = emb[(((size_t)b * kSEQ) + (kSEQ - kINJ) + (k >> 7)) * kD + (k & 127)];
  }
  __syncthreads();
  const int j = blockIdx.x * 256 + t;        // 0..16383
  float acc[kB];
  const float bj = bc[j];
#pragma unroll
  for (int b = 0; b < kB; ++b) acc[b] = bj;
  for (int k = 0; k < kINJ * kD; ++k) {
    const float w = Wc[(size_t)k * (kNB * kD) + j];
#pragma unroll
    for (int b = 0; b < kB; ++b) acc[b] += lsh[b * 4096 + k] * w;
  }
#pragma unroll
  for (int b = 0; b < kB; ++b) proj[(size_t)b * (kNB * kD) + j] = acc[b];
}

// write proj into states[0,b,0:NB,:] (f32) and sT[0,b,:,0:NB] (bf16 transposed)
__global__ void k_scatter_proj(const float* __restrict__ proj,
                               float* __restrict__ S, __bf16* __restrict__ sT)
{
  const int b = blockIdx.y, n = blockIdx.x, d = threadIdx.x;
  const float v = proj[((size_t)b * kNB + n) * kD + d];
  S [(((size_t)b) * kN + n) * kD + d] = v;
  sT[(((size_t)b) * kD + d) * kN + n] = (__bf16)v;
}

// ---- two-stage column reductions: block (128,4) ----
// part[mat*C + c][d] = sum over chunk rows of src[mat][rowskip + c*chunkRows + r][d]
__global__ void k_rpart(const float* __restrict__ src, float* __restrict__ part,
                        int chunkRows, int rowskip, size_t matstride, int C)
{
  __shared__ float red[4][128];
  const int mat = blockIdx.x, c = blockIdx.y;
  const int d = threadIdx.x, ty = threadIdx.y;
  const float* p = src + (size_t)mat * matstride +
                   ((size_t)rowskip + (size_t)c * chunkRows) * kD + d;
  float a = 0.f;
  for (int r = ty; r < chunkRows; r += 4) a += p[(size_t)r * kD];
  red[ty][d] = a;
  __syncthreads();
  if (ty == 0)
    part[((size_t)mat * C + c) * kD + d] =
        red[0][d] + red[1][d] + red[2][d] + red[3][d];
}

// weighted partial: part[b*C + c][d] = sum_s attn[b,s] * emb[b,s,d] over chunk
__global__ void k_wpart(const float* __restrict__ emb,
                        const float* __restrict__ attn,
                        float* __restrict__ part, int C)
{
  __shared__ float red[4][128];
  const int b = blockIdx.x, c = blockIdx.y;
  const int d = threadIdx.x, ty = threadIdx.y;
  const int chunk = kSEQ / 16;
  const int s0 = c * chunk;
  float a = 0.f;
  for (int s = ty; s < chunk; s += 4)
    a += attn[b * kSEQ + s0 + s] * emb[((size_t)b * kSEQ + s0 + s) * kD + d];
  red[ty][d] = a;
  __syncthreads();
  if (ty == 0)
    part[((size_t)b * C + c) * kD + d] =
        red[0][d] + red[1][d] + red[2][d] + red[3][d];
}

__global__ void k_rfin(const float* __restrict__ part, float* __restrict__ dst,
                       int C, float scale)
{
  const int mat = blockIdx.x, d = threadIdx.x;
  float a = 0.f;
  for (int c = 0; c < C; ++c) a += part[((size_t)mat * C + c) * kD + d];
  dst[mat * kD + d] = a * scale;
}

// inj[b,e] = bhin[e] + sum_{z,d} buf[z,b,d] * Whin[z*128+d, e]
__global__ void k_inj(const float* __restrict__ buf,
                      const float* __restrict__ Whin,
                      const float* __restrict__ bhin, float* __restrict__ inj)
{
  const int b = blockIdx.x, e = threadIdx.x;
  float a = bhin[e];
  for (int z = 0; z < kZ; ++z)
    for (int d = 0; d < kD; ++d)
      a += buf[(z * kB + b) * kD + d] * Whin[((z * kD) + d) * kD + e];
  inj[b * kD + e] = a;
}

// g[z,b,e] = tanh(sum_d hub_out[b,d] * Wzh[z,d,e])
__global__ void k_g(const float* __restrict__ hub_out,
                    const float* __restrict__ Wzh, float* __restrict__ g)
{
  const int z = blockIdx.y, b = blockIdx.x, e = threadIdx.x;
  float a = 0.f;
  for (int d = 0; d < kD; ++d)
    a += hub_out[b * kD + d] * Wzh[((size_t)z * kD + d) * kD + e];
  g[(z * kB + b) * kD + e] = tanhf(a);
}

// fused: S[zb, n, d] += g[zb, d] (if g); repack transposed bf16 T[zb, d, n]
__global__ __launch_bounds__(256) void k_post(
    float* __restrict__ S, const float* __restrict__ g,
    __bf16* __restrict__ T, int rows)
{
  __shared__ float tile[64][65];
  const int zb = blockIdx.y, n0 = blockIdx.x * 64, t = threadIdx.x;
  float* Sm       = S + (size_t)zb * rows * kD;
  __bf16* Tm      = T + (size_t)zb * kD * rows;
  const float* gv = g ? (g + zb * kD) : nullptr;
  for (int i = t; i < 64 * kD; i += 256) {
    const int n = i >> 7, d = i & 127;
    float v = Sm[(size_t)(n0 + n) * kD + d];
    if (gv) { v += gv[d]; Sm[(size_t)(n0 + n) * kD + d] = v; }
    tile[n][d] = v;
  }
  __syncthreads();
  for (int i = t; i < 64 * kD; i += 256) {
    const int d = i >> 6, n = i & 63;
    Tm[(size_t)d * rows + n0 + n] = (__bf16)tile[n][d];
  }
}

// qt[b,d] = sum_e Wk[d,e] * q[b,e];  c0[b] = q[b,:].bk   (q = z5@Wq+bq in LDS)
__global__ void k_qproj(const float* __restrict__ z5, const float* __restrict__ Wq,
                        const float* __restrict__ bq, const float* __restrict__ Wk,
                        const float* __restrict__ bk, float* __restrict__ qt,
                        float* __restrict__ c0)
{
  __shared__ float qs[kD];
  const int b = blockIdx.x, e = threadIdx.x;
  float a = bq[e];
  for (int d = 0; d < kD; ++d) a += z5[b * kD + d] * Wq[d * kD + e];
  qs[e] = a;
  __syncthreads();
  float tq = 0.f;
  for (int j = 0; j < kD; ++j) tq += Wk[e * kD + j] * qs[j];
  qt[b * kD + e] = tq;
  if (e == 0) {
    float s = 0.f;
    for (int j = 0; j < kD; ++j) s += qs[j] * bk[j];
    c0[b] = s;
  }
}

// scores[b,s] = (emb[b,s,:].qt[b,:] + c0[b]) / sqrt(D)    one wave per s
__global__ __launch_bounds__(256) void k_scores(
    const float* __restrict__ emb, const float* __restrict__ qt,
    const float* __restrict__ c0, float* __restrict__ scores)
{
  const int b = blockIdx.y;
  const int s = blockIdx.x * 8 + (threadIdx.x >> 5);
  const int lane = threadIdx.x & 31;
  const float* er = emb + ((size_t)b * kSEQ + s) * kD;
  const float* qr = qt + b * kD;
  float a = 0.f;
#pragma unroll
  for (int i = 0; i < 4; ++i) a += er[lane + 32 * i] * qr[lane + 32 * i];
#pragma unroll
  for (int off = 16; off > 0; off >>= 1) a += __shfl_xor(a, off, 32);
  if (lane == 0)
    scores[b * kSEQ + s] = (a + c0[b]) * 0.08838834764831845f;  // 1/sqrt(128)
}

__global__ __launch_bounds__(256) void k_smstats(const float* __restrict__ scores,
                                                 float* __restrict__ stats)
{
  __shared__ float red[256];
  const int b = blockIdx.x, t = threadIdx.x;
  float m = -1e30f;
  for (int s = t; s < kSEQ; s += 256) m = fmaxf(m, scores[b * kSEQ + s]);
  red[t] = m; __syncthreads();
  for (int o = 128; o > 0; o >>= 1) {
    if (t < o) red[t] = fmaxf(red[t], red[t + o]);
    __syncthreads();
  }
  m = red[0]; __syncthreads();
  float se = 0.f;
  for (int s = t; s < kSEQ; s += 256) se += expf(scores[b * kSEQ + s] - m);
  red[t] = se; __syncthreads();
  for (int o = 128; o > 0; o >>= 1) {
    if (t < o) red[t] += red[t + o];
    __syncthreads();
  }
  if (t == 0) { stats[2 * b] = m; stats[2 * b + 1] = red[0]; }
}

__global__ void k_attnw(float* __restrict__ scores, const float* __restrict__ stats)
{
  const int b = blockIdx.x;
  const float m = stats[2 * b], inv = 1.f / stats[2 * b + 1];
  for (int s = threadIdx.x; s < kSEQ; s += blockDim.x)
    scores[b * kSEQ + s] = expf(scores[b * kSEQ + s] - m) * inv;
}

__global__ void k_ctx(const float* __restrict__ wemb, const float* __restrict__ Wv,
                      const float* __restrict__ bv, float* __restrict__ ctx)
{
  const int b = blockIdx.x, e = threadIdx.x;
  float a = bv[e];
  for (int d = 0; d < kD; ++d) a += wemb[b * kD + d] * Wv[d * kD + e];
  ctx[b * kD + e] = a;
}

__global__ void k_comb(const float* __restrict__ z5, const float* __restrict__ ctx,
                       const float* __restrict__ Wcb, const float* __restrict__ bcb,
                       float* __restrict__ comb)
{
  const int b = blockIdx.x, o = threadIdx.x;
  float a = bcb[o];
  for (int d = 0; d < kD; ++d)      a += z5 [b * kD + d] * Wcb[d * kD + o];
  for (int d = 0; d < kD; ++d)      a += ctx[b * kD + d] * Wcb[(kD + d) * kD + o];
  comb[b * kD + o] = a;
}

__global__ __launch_bounds__(256) void k_logits(
    const float* __restrict__ comb, const float* __restrict__ Wo,
    const float* __restrict__ bo, float* __restrict__ out)
{
  const int j = blockIdx.x * 256 + threadIdx.x;
  if (j >= kV) return;
  float acc[kB];
  const float bj = bo[j];
#pragma unroll
  for (int b = 0; b < kB; ++b) acc[b] = bj;
  for (int d = 0; d < kD; ++d) {
    const float w = Wo[(size_t)d * kV + j];
#pragma unroll
    for (int b = 0; b < kB; ++b) acc[b] += comb[b * kD + d] * w;
  }
#pragma unroll
  for (int b = 0; b < kB; ++b) out[(size_t)b * kV + j] = acc[b];
}

// ---------------------------------------------------------------------------
extern "C" void kernel_launch(void* const* d_in, const int* in_sizes, int n_in,
                              void* d_out, int out_size, void* d_ws, size_t ws_size,
                              hipStream_t stream)
{
  (void)in_sizes; (void)n_in; (void)out_size; (void)ws_size;
  const int*   input_ids = (const int*)  d_in[0];
  const float* tok_emb   = (const float*)d_in[1];
  const float* pos_emb   = (const float*)d_in[2];
  const float* Wc        = (const float*)d_in[3];
  const float* bc        = (const float*)d_in[4];
  const float* Wz        = (const float*)d_in[5];
  const float* Wzh       = (const float*)d_in[6];
  const float* Whin      = (const float*)d_in[7];
  const float* bhin      = (const float*)d_in[8];
  const float* Wh        = (const float*)d_in[9];
  const float* Wq        = (const float*)d_in[10];
  const float* bq        = (const float*)d_in[11];
  const float* Wk        = (const float*)d_in[12];
  const float* bk        = (const float*)d_in[13];
  const float* Wv        = (const float*)d_in[14];
  const float* bv        = (const float*)d_in[15];
  const float* Wcb       = (const float*)d_in[16];
  const float* bcb       = (const float*)d_in[17];
  const float* Wo        = (const float*)d_in[18];
  const float* bo        = (const float*)d_in[19];
  float* out = (float*)d_out;

  // ---- workspace carve-up (256B aligned) ----
  char* base = (char*)d_ws;
  size_t off = 0;
  auto alloc = [&](size_t bytes) {
    char* p = base + off;
    off = (off + bytes + 255) & ~(size_t)255;
    return p;
  };
  float*  emb     = (float*) alloc((size_t)kB * kSEQ * kD * 4);        // 8 MB
  float*  states  = (float*) alloc((size_t)kZ * kB * kN * kD * 4);     // 40 MB
  __bf16* sT      = (__bf16*)alloc((size_t)kZ * kB * kD * kN * 2);     // 20 MB
  __bf16* Wzb     = (__bf16*)alloc((size_t)kZ * kN * kN * 2);          // 40 MB
  float*  hub     = (float*) alloc((size_t)kB * kH * kD * 4);          // 4 MB
  __bf16* hT      = (__bf16*)alloc((size_t)kB * kD * kH * 2);          // 2 MB
  __bf16* Whb     = (__bf16*)alloc((size_t)kH * kH * 2);               // 2 MB
  float*  proj    = (float*) alloc((size_t)kB * kNB * kD * 4);
  float*  buf     = (float*) alloc((size_t)kZ * kB * kD * 4);
  float*  inj     = (float*) alloc((size_t)kB * kD * 4);
  float*  hub_out = (float*) alloc((size_t)kB * kD * 4);
  float*  g       = (float*) alloc((size_t)kZ * kB * kD * 4);
  float*  z5      = (float*) alloc((size_t)kB * kD * 4);
  float*  qt      = (float*) alloc((size_t)kB * kD * 4);
  float*  c0      = (float*) alloc((size_t)kB * 4);
  float*  scores  = (float*) alloc((size_t)kB * kSEQ * 4);
  float*  stats   = (float*) alloc((size_t)kB * 2 * 4);
  float*  wemb    = (float*) alloc((size_t)kB * kD * 4);
  float*  ctx     = (float*) alloc((size_t)kB * kD * 4);
  float*  comb    = (float*) alloc((size_t)kB * kD * 4);
  float*  part    = (float*) alloc((size_t)kZ * kB * 16 * kD * 4);     // partials

  // ---- weight conversion to bf16 ----
  {
    const size_t nz = (size_t)kZ * kN * kN;
    k_f2bf<<<(unsigned)((nz + 255) / 256), 256, 0, stream>>>(Wz, Wzb, nz);
    const size_t nh = (size_t)kH * kH;
    k_f2bf<<<(unsigned)((nh + 255) / 256), 256, 0, stream>>>(Wh, Whb, nh);
  }

  // ---- embeddings + inject projection ----
  k_embed<<<dim3(kSEQ, kB), kD, 0, stream>>>(input_ids, tok_emb, pos_emb, emb);
  k_inject<<<(kNB * kD) / 256, 256, (size_t)kB * kINJ * kD * 4, stream>>>(
      emb, Wc, bc, proj);

  // ---- initial state ----
  hipMemsetAsync(states, 0, (size_t)kZ * kB * kN * kD * 4, stream);
  hipMemsetAsync(sT,     0, (size_t)kZ * kB * kD * kN * 2, stream);
  hipMemsetAsync(hub,    0, (size_t)kB * kH * kD * 4, stream);
  hipMemsetAsync(hT,     0, (size_t)kB * kD * kH * 2, stream);
  k_scatter_proj<<<dim3(kNB, kB), kD, 0, stream>>>(proj, states, sT);

  const dim3 blk2(128, 4);

  // ---- wave propagation ----
  for (int w = 0; w < kWAVES; ++w) {
    k_zone_gemm<<<dim3(kN / 128, kB, kZ), 256, 0, stream>>>(Wzb, sT, states);
    // buf[z,b,:] = mean of last NB rows
    k_rpart<<<dim3(kZ * kB, 1), blk2, 0, stream>>>(states, part, kNB, kN - kNB,
                                                   (size_t)kN * kD, 1);
    k_rfin<<<kZ * kB, kD, 0, stream>>>(part, buf, 1, 1.f / kNB);
    k_inj<<<kB, kD, 0, stream>>>(buf, Whin, bhin, inj);
    k_hub_gemm<<<dim3(kH / 128, kB), 256, 0, stream>>>(Whb, hT, inj, hub);
    // hub_out[b,:] = mean over H rows
    k_rpart<<<dim3(kB, 8), blk2, 0, stream>>>(hub, part, kH / 8, 0,
                                              (size_t)kH * kD, 8);
    k_rfin<<<kB, kD, 0, stream>>>(part, hub_out, 8, 1.f / kH);
    k_post<<<dim3(kH / 64, kB), 256, 0, stream>>>(hub, nullptr, hT, kH);
    k_g<<<dim3(kB, kZ), kD, 0, stream>>>(hub_out, Wzh, g);
    k_post<<<dim3(kN / 64, kZ * kB), 256, 0, stream>>>(states, g, sT, kN);
  }

  // ---- output head ----
  k_rpart<<<dim3(kB, 16), blk2, 0, stream>>>(
      states + (size_t)(kZ - 1) * kB * kN * kD, part, kN / 16, 0,
      (size_t)kN * kD, 16);
  k_rfin<<<kB, kD, 0, stream>>>(part, z5, 16, 1.f / kN);
  k_qproj<<<kB, kD, 0, stream>>>(z5, Wq, bq, Wk, bk, qt, c0);
  k_scores<<<dim3(kSEQ / 8, kB), 256, 0, stream>>>(emb, qt, c0, scores);
  k_smstats<<<kB, 256, 0, stream>>>(scores, stats);
  k_attnw<<<kB, 256, 0, stream>>>(scores, stats);
  k_wpart<<<dim3(kB, 16), blk2, 0, stream>>>(emb, scores, part, 16);
  k_rfin<<<kB, kD, 0, stream>>>(part, wemb, 16, 1.f);
  k_ctx<<<kB, kD, 0, stream>>>(wemb, Wv, bv, ctx);
  k_comb<<<kB, kD, 0, stream>>>(z5, ctx, Wcb, bcb, comb);
  k_logits<<<(kV + 255) / 256, 256, 0, stream>>>(comb, Wo, bo, out);
}
